// BaseConvBlock_31834297598029
// MI455X (gfx1250) — matrix operand order repaired and checked by reading, not compiled, not run on hardware
//
#include <hip/hip_runtime.h>

typedef __attribute__((ext_vector_type(16))) __bf16       v16bf;
typedef __attribute__((ext_vector_type(8)))  __bf16       v8bf;
typedef __attribute__((ext_vector_type(8)))  float        v8f;
typedef __attribute__((ext_vector_type(4)))  float        f32x4;
typedef __attribute__((ext_vector_type(4)))  unsigned int u32x4;

#define NPTS        131072
#define CIN         64
#define COUT        64
#define KOFF        27
#define NWG         256
#define NTHREADS    512                         // 16 waves
#define PTS_PER_WG  (NPTS / NWG)                // 512
#define ROWS_PER_IT 128                         // slab: 8 row-tiles of 16
#define ITERS       (PTS_PER_WG / ROWS_PER_IT)  // 4

#define SW_ELEMS  (KOFF * CIN * COUT)           // 110592 ushorts = 221184 B
#define SA_ELEMS  (ROWS_PER_IT * CIN)           // 8192 ushorts = 16384 B per buffer
#define LDS_BYTES (SW_ELEMS * 2 + 2 * SA_ELEMS * 2 + NTHREADS * 4)  // 256000 B

// Native bf16 converts (v_cvt_*bf16* on CDNA5) instead of manual integer RNE.
__device__ __forceinline__ u32x4 pack_bf16x8(f32x4 a, f32x4 b) {
  v8bf v;
  v[0] = (__bf16)a[0]; v[1] = (__bf16)a[1]; v[2] = (__bf16)a[2]; v[3] = (__bf16)a[3];
  v[4] = (__bf16)b[0]; v[5] = (__bf16)b[1]; v[6] = (__bf16)b[2]; v[7] = (__bf16)b[3];
  return __builtin_bit_cast(u32x4, v);
}

// ---------------- Kernel 0: W fp32 [k][cin][cout] -> bf16 transposed [k][cout][cin]
__global__ void wt_convert(const float* __restrict__ W,
                           unsigned short* __restrict__ Wt) {
  int i = blockIdx.x * 256 + threadIdx.x;   // grid sized exactly: SW_ELEMS threads
  int k = i >> 12;
  int r = i & 4095;
  int d = r >> 6;             // cout
  int c = r & 63;             // cin
  __bf16 h = (__bf16)W[(k << 12) + (c << 6) + d];
  Wt[i] = __builtin_bit_cast(unsigned short, h);
}

// ---------------- Kernel 1: gather + WMMA GEMM over 27 offsets, raw y + BN partials
__global__ __launch_bounds__(NTHREADS)
void spconv_wmma(const float* __restrict__ x,
                 const unsigned short* __restrict__ Wt,
                 const int* __restrict__ nidx,
                 const unsigned char* __restrict__ mask,
                 float* __restrict__ y,          // raw y into d_out
                 float* __restrict__ partial) {  // [NWG][128]: 64 sum + 64 sumsq
  extern __shared__ char smem[];
  unsigned short* sW   = (unsigned short*)smem;                            // 221184 B
  unsigned short* sA   = (unsigned short*)(smem + SW_ELEMS * 2);           // 2 x 16384 B
  float*          sRed = (float*)(smem + SW_ELEMS * 2 + 2 * SA_ELEMS * 2); // 2048 B

  const int tid  = threadIdx.x;
  const int wg   = blockIdx.x;
  const int wave = tid >> 5;
  const int lane = tid & 31;
  const int couttile = wave & 3;            // 0..3 (16 cols of Cout=64)
  const int T0 = (wave >> 2) * 2;           // row-tile pair 0..7
  const int T1 = T0 + 1;
  const int half = lane >> 4;               // 0..1
  const int l15  = lane & 15;
  const int seg  = tid & 7;                 // 8-float segment of a row
  const int r0   = tid >> 3;                // rows r0 and r0+64 per thread

  // --- async DMA of bf16 W into LDS (ASYNCcnt-tracked), overlapped with first gather
  {
    unsigned lds_w = (unsigned)(unsigned long long)(void*)sW;
    unsigned long long gbase = (unsigned long long)Wt;
    for (int i = tid; i < SW_ELEMS / 8; i += NTHREADS) {   // 27 chunks/thread exactly
      unsigned ldsa = lds_w + (unsigned)(i * 16);
      unsigned long long ga = gbase + (unsigned long long)(unsigned)(i * 16);
      asm volatile("global_load_async_to_lds_b128 %0, %1, off"
                   :: "v"(ldsa), "v"(ga) : "memory");
    }
  }

  float bnsum = 0.0f, bnsq = 0.0f;

  for (int it = 0; it < ITERS; ++it) {
    const int sb = wg * PTS_PER_WG + it * ROWS_PER_IT;   // slab base row
    v8f acc0 = {0.f, 0.f, 0.f, 0.f, 0.f, 0.f, 0.f, 0.f};
    v8f acc1 = {0.f, 0.f, 0.f, 0.f, 0.f, 0.f, 0.f, 0.f};

    // ---- prologue: stage k=0 into buf0, prefetch idx/mask for k=1
    {
      const int n0 = sb + r0, n1 = sb + r0 + 64;
      const int i0 = nidx[n0], i1 = nidx[n1];
      const int m0 = mask[n0], m1 = mask[n1];
      f32x4 a0 = {0.f,0.f,0.f,0.f}, a1 = a0, b0 = a0, b1 = a0;
      if (m0) { const f32x4* p = (const f32x4*)(x + (i0 << 6) + (seg << 3)); a0 = p[0]; a1 = p[1]; }
      if (m1) { const f32x4* p = (const f32x4*)(x + (i1 << 6) + (seg << 3)); b0 = p[0]; b1 = p[1]; }
      *(u32x4*)(sA + (r0 << 6) + (seg << 3))        = pack_bf16x8(a0, a1);
      *(u32x4*)(sA + ((r0 + 64) << 6) + (seg << 3)) = pack_bf16x8(b0, b1);
    }
    int idn0 = nidx[NPTS + sb + r0], idn1 = nidx[NPTS + sb + r0 + 64];
    int mn0  = mask[NPTS + sb + r0], mn1  = mask[NPTS + sb + r0 + 64];

    asm volatile("s_wait_asynccnt 0x0" ::: "memory");  // W resident (free after it=0)
    __syncthreads();

    for (int k = 0; k < KOFF; ++k) {
      const unsigned short* bufC = sA + (k & 1) * SA_ELEMS;
      unsigned short*       bufN = sA + ((k + 1) & 1) * SA_ELEMS;

      // --- issue gathers for k+1 (hidden under WMMAs below)
      f32x4 g0a = {0.f,0.f,0.f,0.f}, g0b = g0a, g1a = g0a, g1b = g0a;
      if (k < KOFF - 1) {
        if (mn0) { const f32x4* p = (const f32x4*)(x + (idn0 << 6) + (seg << 3)); g0a = p[0]; g0b = p[1]; }
        if (mn1) { const f32x4* p = (const f32x4*)(x + (idn1 << 6) + (seg << 3)); g1a = p[0]; g1b = p[1]; }
      }
      // --- prefetch idx/mask for k+2
      int idf0 = idn0, idf1 = idn1, mf0 = mn0, mf1 = mn1;
      if (k < KOFF - 2) {
        const long long o = (long long)(k + 2) * NPTS + sb;
        idf0 = nidx[o + r0]; idf1 = nidx[o + r0 + 64];
        mf0  = mask[o + r0]; mf1  = mask[o + r0 + 64];
      }

      // --- 4 WMMAs: two row-tiles x two cin chunks, shared B fragment per chunk
      // A lane layout: row M = l15, K = c*32 + (e<8?0:16) + half*8 + (e&7)
      // B lane layout: col N = l15, K = c*32 + (e<8?0:16) + half*8 + (e&7)
      const unsigned short* aR0 = bufC + ((T0 * 16 + l15) << 6) + (half << 3);
      const unsigned short* aR1 = bufC + ((T1 * 16 + l15) << 6) + (half << 3);
      const unsigned short* bC  = sW + (k << 12) + ((couttile * 16 + l15) << 6) + (half << 3);
#pragma unroll
      for (int c = 0; c < 2; ++c) {
        union { u32x4 q[2]; v16bf v; } ub, ua0, ua1;
        ub.q[0]  = *(const u32x4*)(bC + c * 32);
        ub.q[1]  = *(const u32x4*)(bC + c * 32 + 16);
        ua0.q[0] = *(const u32x4*)(aR0 + c * 32);
        ua0.q[1] = *(const u32x4*)(aR0 + c * 32 + 16);
        ua1.q[0] = *(const u32x4*)(aR1 + c * 32);
        ua1.q[1] = *(const u32x4*)(aR1 + c * 32 + 16);
        acc0 = __builtin_amdgcn_wmma_f32_16x16x32_bf16(
            false, ua0.v, false, ub.v, (short)0, acc0, false, false);
        acc1 = __builtin_amdgcn_wmma_f32_16x16x32_bf16(
            false, ua1.v, false, ub.v, (short)0, acc1, false, false);
      }

      // --- convert + store k+1 tile into the other buffer; rotate idx pipeline
      if (k < KOFF - 1) {
        *(u32x4*)(bufN + (r0 << 6) + (seg << 3))        = pack_bf16x8(g0a, g0b);
        *(u32x4*)(bufN + ((r0 + 64) << 6) + (seg << 3)) = pack_bf16x8(g1a, g1b);
        idn0 = idf0; idn1 = idf1; mn0 = mf0; mn1 = mf1;
      }
      __syncthreads();   // single barrier per k (double-buffered)
    }

    // --- store raw y tiles + accumulate BN partials
    // C layout: VGPR v -> row M = v + half*8, col N = l15 (per 16-lane half)
    const int col  = couttile * 16 + l15;
    const int rowA = sb + T0 * 16 + half * 8;
    const int rowB = sb + T1 * 16 + half * 8;
#pragma unroll
    for (int v = 0; v < 8; ++v) {
      float va = acc0[v], vb = acc1[v];
      y[(rowA + v) * COUT + col] = va;
      y[(rowB + v) * COUT + col] = vb;
      bnsum += va + vb;
      bnsq  += va * va + vb * vb;
    }
  }

  // --- workgroup BN partial reduction (deterministic, no atomics)
  sRed[tid] = bnsum;
  __syncthreads();
  if (tid < 64) {
    const int ct = tid >> 4, l = tid & 15;
    float s = 0.f;
    for (int w = ct; w < 16; w += 4)
      s += sRed[w * 32 + l] + sRed[w * 32 + 16 + l];
    partial[wg * 128 + tid] = s;
  }
  __syncthreads();
  sRed[tid] = bnsq;
  __syncthreads();
  if (tid < 64) {
    const int ct = tid >> 4, l = tid & 15;
    float s = 0.f;
    for (int w = ct; w < 16; w += 4)
      s += sRed[w * 32 + l] + sRed[w * 32 + 16 + l];
    partial[wg * 128 + 64 + tid] = s;
  }
}

// ---------------- Kernel 2: final BN stats -> per-channel scale/shift
__global__ void bn_stats(const float* __restrict__ partial,
                         const float* __restrict__ gamma,
                         const float* __restrict__ beta,
                         float* __restrict__ ss) {
  int c = threadIdx.x;
  if (c >= COUT) return;
  float s = 0.f, q = 0.f;
  for (int g = 0; g < NWG; ++g) {
    s += partial[g * 128 + c];
    q += partial[g * 128 + 64 + c];
  }
  const float inv_n = 1.0f / (float)NPTS;
  float mean = s * inv_n;
  float var  = q * inv_n - mean * mean;
  float scale = gamma[c] * rsqrtf(var + 1e-5f);
  ss[c]      = scale;
  ss[64 + c] = beta[c] - mean * scale;
}

// ---------------- Kernel 3: in-place BN apply + ReLU (float4 vectorized)
__global__ void bn_apply(float* __restrict__ y, const float* __restrict__ ss) {
  int i = blockIdx.x * 256 + threadIdx.x;   // one float4 per thread
  f32x4* p = (f32x4*)y;
  f32x4 v = p[i];
  int col = (i & 15) << 2;                  // (i*4) & 63
  const f32x4 sc = *(const f32x4*)(ss + col);
  const f32x4 sh = *(const f32x4*)(ss + 64 + col);
#pragma unroll
  for (int j = 0; j < 4; ++j) v[j] = fmaxf(fmaf(v[j], sc[j], sh[j]), 0.0f);
  p[i] = v;
}

extern "C" void kernel_launch(void* const* d_in, const int* in_sizes, int n_in,
                              void* d_out, int out_size, void* d_ws, size_t ws_size,
                              hipStream_t stream) {
  const float* x          = (const float*)d_in[0];
  const float* W          = (const float*)d_in[1];
  const float* gamma      = (const float*)d_in[2];
  const float* beta       = (const float*)d_in[3];
  const int* nidx         = (const int*)d_in[4];
  const unsigned char* mk = (const unsigned char*)d_in[5];  // jnp.bool_ = 1 byte
  float* out = (float*)d_out;

  char* ws = (char*)d_ws;
  unsigned short* Wt = (unsigned short*)ws;                        // 221184 B
  float* partial     = (float*)(ws + SW_ELEMS * 2);                // 131072 B
  float* ss          = (float*)(ws + SW_ELEMS * 2 + NWG * 128 * 4);// 128 floats

  // Allow 250 KB dynamic LDS (WGP has 320 KB); deterministic, capture-safe.
  (void)hipFuncSetAttribute((const void*)spconv_wmma,
                            hipFuncAttributeMaxDynamicSharedMemorySize,
                            LDS_BYTES);

  wt_convert<<<SW_ELEMS / 256, 256, 0, stream>>>(W, Wt);
  spconv_wmma<<<NWG, NTHREADS, LDS_BYTES, stream>>>(x, Wt, nidx, mk, out, partial);
  bn_stats<<<1, 64, 0, stream>>>(partial, gamma, beta, ss);
  bn_apply<<<(NPTS * COUT / 4) / 256, 256, 0, stream>>>(out, ss);
}
